// Transformer_14980845929019
// MI455X (gfx1250) — compile-verified, matrix operand order checked by reading, not run on hardware
//
#include <hip/hip_runtime.h>
#include <cstdint>
#include <cstddef>

// ---------------- model constants (match reference) ----------------
#define D_MODEL 1024
#define N_HEAD  16
#define D_HEAD  64
#define D_FF    4096
#define N_LAYER 4
#define VOCAB   32000
#define NB      4
#define SEQ     512
#define LN_EPS  1e-5f

typedef unsigned short u16;

// ---------------- WMMA types ----------------
typedef __attribute__((ext_vector_type(16))) __bf16 v16bf;
typedef __attribute__((ext_vector_type(8)))  float  v8f;

struct __align__(16) U4 { unsigned x, y, z, w; };
union Frag { U4 q[2]; v16bf v; };
struct __align__(16) H8 { u16 h[8]; };

__device__ __forceinline__ u16 f2bf(float f) {
  // round-to-nearest-even fp32 -> bf16
  unsigned u = __float_as_uint(f);
  unsigned r = 0x7FFFu + ((u >> 16) & 1u);
  return (u16)((u + r) >> 16);
}

// =====================================================================
// Generic strided-batched bf16 GEMM via WMMA, double-buffered LDS fed by
// async global->LDS copies (ASYNCcnt-pipelined).
//   C[m,n] = act( alpha * sum_k A[m,k]*B[n,k] + bias[n] )      (B is [N][K])
//   outputs: optional fp32 C, optional bf16 shadow (row-major or transposed)
// =====================================================================
struct GemmBF {
  const u16* A;        // bf16 [M][K] (lda)
  const u16* B;        // bf16 [N][K] (ldb)  -- always K-contiguous
  const float* bias;   // fp32 [N] or null
  float* C;            // fp32 [M][N] (ldc) or null
  u16* Csh;            // bf16 shadow or null
  int M, N, K;
  int lda, ldb, ldc, ldsh;
  long long sAo, sAi, sBo, sBi, sCo, sCi, sSo, sSi;
  int innerB;
  float alpha;
  int relu;
  int shMode;          // 0 none, 1 row-major [M][N], 2 transposed [N][M]
  int writeC;
};

#define BK 32
#define LDS_AS 40   // padded LDS row stride in bf16 (80B rows; 16B-aligned segs)

template<int WM, int WN, int TM, int TN>   // waves WM x WN, wave tile (TM*16)x(TN*16)
__global__ __launch_bounds__(256)
void gemm_bf16(GemmBF p) {
  constexpr int BMt = WM * TM * 16;
  constexpr int BNt = WN * TN * 16;
  constexpr int NC  = BMt / 64 + BNt / 64;      // async copies per thread per tile
  __shared__ __align__(16) u16 As[2][BMt][LDS_AS];   // [buf][m][k]
  __shared__ __align__(16) u16 Bs[2][BNt][LDS_AS];   // [buf][n][k]

  const int tid  = threadIdx.x;
  const int wave = tid >> 5;
  const int lane = tid & 31;
  const int l16  = lane & 15;
  const int hi   = lane >> 4;
  const int waveM = wave / WN;
  const int waveN = wave % WN;

  const int blockN = blockIdx.x * BNt;
  const int blockM = blockIdx.y * BMt;
  const int z  = blockIdx.z;
  const int zo = z / p.innerB;
  const int zi = z % p.innerB;

  const u16* Ab = p.A + (long long)zo * p.sAo + (long long)zi * p.sAi;
  const u16* Bb = p.B + (long long)zo * p.sBo + (long long)zi * p.sBi;

  // LDS byte offsets for async destinations (low 32 bits of LDS aperture addr)
  const unsigned asBase = (unsigned)(uintptr_t)&As[0][0][0];
  const unsigned bsBase = (unsigned)(uintptr_t)&Bs[0][0][0];

  // stage one BK-slice into LDS buffer `buf` with async copies (16B units)
  auto stage = [&](int k0, int buf) {
#pragma unroll
    for (int i = 0; i < BMt / 64; ++i) {
      int u = tid + i * 256;
      int m = u >> 2, seg = u & 3;
      const u16* g = Ab + (long long)(blockM + m) * p.lda + (k0 + seg * 8);
      unsigned l = asBase + (unsigned)(((buf * BMt + m) * LDS_AS + seg * 8) * 2);
      asm volatile("global_load_async_to_lds_b128 %0, %1, off"
                   :: "v"(l), "v"(g) : "memory");
    }
#pragma unroll
    for (int i = 0; i < BNt / 64; ++i) {
      int u = tid + i * 256;
      int n = u >> 2, seg = u & 3;
      const u16* g = Bb + (long long)(blockN + n) * p.ldb + (k0 + seg * 8);
      unsigned l = bsBase + (unsigned)(((buf * BNt + n) * LDS_AS + seg * 8) * 2);
      asm volatile("global_load_async_to_lds_b128 %0, %1, off"
                   :: "v"(l), "v"(g) : "memory");
    }
  };

  const v8f vzero = {0.f,0.f,0.f,0.f,0.f,0.f,0.f,0.f};
  v8f acc[TM][TN];
#pragma unroll
  for (int tm = 0; tm < TM; ++tm)
#pragma unroll
    for (int tn = 0; tn < TN; ++tn) acc[tm][tn] = vzero;

  stage(0, 0);                                   // prologue: fill buffer 0
  int cur = 0;
  for (int k0 = 0; k0 < p.K; k0 += BK) {
    const bool more = (k0 + BK) < p.K;
    if (more) {
      // overwrite the buffer whose readers all passed last iteration's end barrier
      stage(k0 + BK, cur ^ 1);
      // in-order async completion: <=NC outstanding => buffer `cur` is resident
      if constexpr (NC == 4)      asm volatile("s_wait_asynccnt 0x4" ::: "memory");
      else if constexpr (NC == 3) asm volatile("s_wait_asynccnt 0x3" ::: "memory");
      else                        asm volatile("s_wait_asynccnt 0x2" ::: "memory");
    } else {
      asm volatile("s_wait_asynccnt 0x0" ::: "memory");
    }
    __syncthreads();                             // all waves' `cur` copies landed

    // ---- fragments per ISA 16-bit 16x32 layout ----
    Frag fa[TM], fb[TN];
#pragma unroll
    for (int tm = 0; tm < TM; ++tm) {
      int m = (waveM * TM + tm) * 16 + l16;
      fa[tm].q[0] = *(const U4*)&As[cur][m][hi * 8];
      fa[tm].q[1] = *(const U4*)&As[cur][m][16 + hi * 8];
    }
#pragma unroll
    for (int tn = 0; tn < TN; ++tn) {
      int n = (waveN * TN + tn) * 16 + l16;
      fb[tn].q[0] = *(const U4*)&Bs[cur][n][hi * 8];
      fb[tn].q[1] = *(const U4*)&Bs[cur][n][16 + hi * 8];
    }
#pragma unroll
    for (int tm = 0; tm < TM; ++tm)
#pragma unroll
      for (int tn = 0; tn < TN; ++tn)
        acc[tm][tn] = __builtin_amdgcn_wmma_f32_16x16x32_bf16(
            false, fa[tm].v, false, fb[tn].v, (short)0, acc[tm][tn], false, false);
    __syncthreads();                             // readers done: `cur` reusable
    cur ^= 1;
  }

  // ---- epilogue ----
  float* Cb = p.writeC ? p.C + (long long)zo * p.sCo + (long long)zi * p.sCi : nullptr;
  u16* Sb = p.shMode ? p.Csh + (long long)zo * p.sSo + (long long)zi * p.sSi : nullptr;
#pragma unroll
  for (int tm = 0; tm < TM; ++tm) {
#pragma unroll
    for (int tn = 0; tn < TN; ++tn) {
      int n  = blockN + (waveN * TN + tn) * 16 + l16;
      int m0 = blockM + (waveM * TM + tm) * 16 + hi * 8;
      float bi = p.bias ? p.bias[n] : 0.0f;
      float vv[8];
#pragma unroll
      for (int j = 0; j < 8; ++j) {
        float v = acc[tm][tn][j] * p.alpha + bi;
        vv[j] = p.relu ? fmaxf(v, 0.0f) : v;
      }
      if (p.writeC) {
#pragma unroll
        for (int j = 0; j < 8; ++j)
          Cb[(long long)(m0 + j) * p.ldc + n] = vv[j];
      }
      if (p.shMode == 1) {
#pragma unroll
        for (int j = 0; j < 8; ++j)
          Sb[(long long)(m0 + j) * p.ldsh + n] = f2bf(vv[j]);
      } else if (p.shMode == 2) {           // transposed: contiguous 16B store
        H8 pk;
#pragma unroll
        for (int j = 0; j < 8; ++j) pk.h[j] = f2bf(vv[j]);
        *(H8*)(Sb + (long long)n * p.ldsh + m0) = pk;
      }
    }
  }
}

// ---------------- weight convert+transpose: fp32 [L][K][N] -> bf16 [L][N][K] ----------------
__global__ __launch_bounds__(256)
void convT(const float* W, u16* Wt, int K, int N) {
  __shared__ float t[32][33];
  long long slice = (long long)blockIdx.z;
  const float* Wz = W + slice * K * N;
  u16* Wtz = Wt + slice * K * N;
  int bx = blockIdx.x * 32;           // N
  int by = blockIdx.y * 32;           // K
  int tx = threadIdx.x & 31, ty = threadIdx.x >> 5;
#pragma unroll
  for (int j = 0; j < 32; j += 8)
    t[ty + j][tx] = Wz[(long long)(by + ty + j) * N + (bx + tx)];
  __syncthreads();
#pragma unroll
  for (int j = 0; j < 32; j += 8)
    Wtz[(long long)(bx + ty + j) * K + (by + tx)] = f2bf(t[tx][ty + j]);
}

// ---------------- embedding + positional encoding (fp32 + bf16 shadow) ----------------
__global__ __launch_bounds__(256)
void embed_pe(const int* tok, const float* emb, float* out, u16* outh,
              int S, int D, float scale) {
  long long idx = (long long)blockIdx.x * 256 + threadIdx.x;
  int d = (int)(idx % D);
  long long bs = idx / D;
  int s = (int)(bs % S);
  int t = tok[bs];
  float ang = (float)s * __expf((float)(d & ~1) * (-9.210340372f / (float)D));
  float pe  = (d & 1) ? __cosf(ang) : __sinf(ang);
  float r = emb[(long long)t * D + d] * scale + pe;
  out[idx]  = r;
  outh[idx] = f2bf(r);
}

// ---------------- masked softmax: fp32 scores in, bf16 probs out; wave32/row ----------------
__global__ __launch_bounds__(256)
void softmax_rows(const float* sc, u16* sch, const int* keyTok, int causal,
                  int H, int Sq, int Sk) {
  long long row = (long long)blockIdx.x * 8 + (threadIdx.x >> 5);
  int lane = threadIdx.x & 31;
  int q = (int)(row % Sq);
  int b = (int)((row / Sq) / H);
  const float* r = sc + row * (long long)Sk;
  u16* o = sch + row * (long long)Sk;
  const int* kt = keyTok ? keyTok + (long long)b * Sk : nullptr;

  float vals[16];                    // Sk == 512 == 32 lanes * 16
  float mx = -INFINITY;
#pragma unroll
  for (int j = 0; j < 16; ++j) {
    int col = lane + 32 * j;
    bool ok = (!causal || col <= q) && (!kt || kt[col] != 0);
    float v = ok ? r[col] : -INFINITY;
    vals[j] = v;
    mx = fmaxf(mx, v);
  }
#pragma unroll
  for (int off = 16; off >= 1; off >>= 1) mx = fmaxf(mx, __shfl_xor(mx, off, 32));
  float sum = 0.f;
#pragma unroll
  for (int j = 0; j < 16; ++j) { float e = __expf(vals[j] - mx); vals[j] = e; sum += e; }
#pragma unroll
  for (int off = 16; off >= 1; off >>= 1) sum += __shfl_xor(sum, off, 32);
  float inv = 1.0f / sum;
#pragma unroll
  for (int j = 0; j < 16; ++j) o[lane + 32 * j] = f2bf(vals[j] * inv);
}

// ---------------- residual add + LayerNorm (fp32 io + bf16 shadow) ----------------
__global__ __launch_bounds__(256)
void add_layernorm(const float* x, const float* rr, const float* g, const float* bb,
                   float* out, u16* sh, int D) {
  __shared__ float red[256];
  int tid = threadIdx.x;
  long long base = (long long)blockIdx.x * D;
  int d0 = tid * 4;                       // D == 1024 == 256*4
  float4 xv = *(const float4*)(x + base + d0);
  float4 rv = *(const float4*)(rr + base + d0);
  float v[4] = { xv.x + rv.x, xv.y + rv.y, xv.z + rv.z, xv.w + rv.w };

  float s = v[0] + v[1] + v[2] + v[3];
  red[tid] = s; __syncthreads();
  for (int off = 128; off >= 1; off >>= 1) { if (tid < off) red[tid] += red[tid + off]; __syncthreads(); }
  float mu = red[0] * (1.0f / D);
  __syncthreads();

  float s2 = 0.f;
#pragma unroll
  for (int j = 0; j < 4; ++j) { float dd = v[j] - mu; s2 += dd * dd; }
  red[tid] = s2; __syncthreads();
  for (int off = 128; off >= 1; off >>= 1) { if (tid < off) red[tid] += red[tid + off]; __syncthreads(); }
  float rstd = rsqrtf(red[0] * (1.0f / D) + LN_EPS);

#pragma unroll
  for (int j = 0; j < 4; ++j) {
    int d = d0 + j;
    float r = g[d] * (v[j] - mu) * rstd + bb[d];
    out[base + d] = r;
    sh[base + d]  = f2bf(r);
  }
}

// =====================================================================
// host-side orchestration
// =====================================================================
extern "C" void kernel_launch(void* const* d_in, const int* in_sizes, int n_in,
                              void* d_out, int out_size, void* d_ws, size_t ws_size,
                              hipStream_t stream) {
  (void)in_sizes; (void)n_in; (void)out_size; (void)ws_size;
  auto fpp = [&](int i) { return (const float*)d_in[i]; };

  const int* src = (const int*)d_in[0];
  const int* tgt = (const int*)d_in[1];
  const float* src_emb = fpp(2);
  const float* tgt_emb = fpp(3);
  const float *eWq=fpp(4),  *ebq=fpp(5),  *eWk=fpp(6),  *ebk=fpp(7);
  const float *eWv=fpp(8),  *ebv=fpp(9),  *eWo=fpp(10), *ebo=fpp(11);
  const float *eW1=fpp(12), *eb1=fpp(13), *eW2=fpp(14), *eb2=fpp(15);
  const float *eln1g=fpp(16), *eln1b=fpp(17), *eln2g=fpp(18), *eln2b=fpp(19);
  const float *dWq=fpp(20), *dbq=fpp(21), *dWk=fpp(22), *dbk=fpp(23);
  const float *dWv=fpp(24), *dbv=fpp(25), *dWo=fpp(26), *dbo=fpp(27);
  const float *cWq=fpp(28), *cbq=fpp(29), *cWk=fpp(30), *cbk=fpp(31);
  const float *cWv=fpp(32), *cbv=fpp(33), *cWo=fpp(34), *cbo=fpp(35);
  const float *dW1=fpp(36), *db1=fpp(37), *dW2=fpp(38), *db2=fpp(39);
  const float *dln1g=fpp(40), *dln1b=fpp(41), *dln2g=fpp(42), *dln2b=fpp(43);
  const float *dln3g=fpp(44), *dln3b=fpp(45);
  const float *outW=fpp(46), *outB=fpp(47);

  const int M = NB * SEQ;                   // 2048 token rows
  uintptr_t ws = (uintptr_t)d_ws;
  auto carve = [&](size_t bytes) {
    void* pptr = (void*)ws;
    ws += (bytes + 255) & ~(size_t)255;
    return pptr;
  };
  // fp32 buffers
  size_t actF = (size_t)M * D_MODEL * sizeof(float);
  float* X  = (float*)carve(actF);
  float* Y  = (float*)carve(actF);
  float* Pb = (float*)carve(actF);
  float* SC = (float*)carve((size_t)NB * N_HEAD * SEQ * SEQ * sizeof(float));
  // bf16 shadows
  size_t actH = (size_t)M * D_MODEL * sizeof(u16);
  u16* Xh  = (u16*)carve(actH);
  u16* Yh  = (u16*)carve(actH);
  u16* Qh  = (u16*)carve(actH);                 // [M][D] row-major
  u16* Kh  = (u16*)carve(actH);                 // [M][D] row-major
  u16* Vt  = (u16*)carve(actH);                 // [D][M] transposed
  u16* AOh = (u16*)carve(actH);
  u16* Hh  = (u16*)carve((size_t)M * D_FF * sizeof(u16));
  u16* SCh = (u16*)carve((size_t)NB * N_HEAD * SEQ * SEQ * sizeof(u16));
  // bf16 transposed weights
  auto wslot = [&](size_t elems) { return (u16*)carve(elems * sizeof(u16)); };
  size_t wDD = (size_t)N_LAYER * D_MODEL * D_MODEL;
  size_t wDF = (size_t)N_LAYER * D_MODEL * D_FF;
  u16 *teWq=wslot(wDD), *teWk=wslot(wDD), *teWv=wslot(wDD), *teWo=wslot(wDD);
  u16 *teW1=wslot(wDF), *teW2=wslot(wDF);
  u16 *tdWq=wslot(wDD), *tdWk=wslot(wDD), *tdWv=wslot(wDD), *tdWo=wslot(wDD);
  u16 *tcWq=wslot(wDD), *tcWk=wslot(wDD), *tcWv=wslot(wDD), *tcWo=wslot(wDD);
  u16 *tdW1=wslot(wDF), *tdW2=wslot(wDF);
  u16 *toutW=wslot((size_t)D_MODEL * VOCAB);

  // ---- convert+transpose all weights: fp32 [L][K][N] -> bf16 [L][N][K] ----
  auto cvt = [&](const float* W, u16* Wt, int K, int N, int L) {
    convT<<<dim3((unsigned)(N / 32), (unsigned)(K / 32), (unsigned)L),
            dim3(256), 0, stream>>>(W, Wt, K, N);
  };
  cvt(eWq, teWq, D_MODEL, D_MODEL, N_LAYER); cvt(eWk, teWk, D_MODEL, D_MODEL, N_LAYER);
  cvt(eWv, teWv, D_MODEL, D_MODEL, N_LAYER); cvt(eWo, teWo, D_MODEL, D_MODEL, N_LAYER);
  cvt(eW1, teW1, D_MODEL, D_FF, N_LAYER);    cvt(eW2, teW2, D_FF, D_MODEL, N_LAYER);
  cvt(dWq, tdWq, D_MODEL, D_MODEL, N_LAYER); cvt(dWk, tdWk, D_MODEL, D_MODEL, N_LAYER);
  cvt(dWv, tdWv, D_MODEL, D_MODEL, N_LAYER); cvt(dWo, tdWo, D_MODEL, D_MODEL, N_LAYER);
  cvt(cWq, tcWq, D_MODEL, D_MODEL, N_LAYER); cvt(cWk, tcWk, D_MODEL, D_MODEL, N_LAYER);
  cvt(cWv, tcWv, D_MODEL, D_MODEL, N_LAYER); cvt(cWo, tcWo, D_MODEL, D_MODEL, N_LAYER);
  cvt(dW1, tdW1, D_MODEL, D_FF, N_LAYER);    cvt(dW2, tdW2, D_FF, D_MODEL, N_LAYER);
  cvt(outW, toutW, D_MODEL, VOCAB, 1);

  // ---- GEMM launchers ----
  auto gemmBig = [&](GemmBF p, int m, int n, int batch) {     // 128x128 tile
    gemm_bf16<2, 4, 4, 2><<<dim3((unsigned)(n / 128), (unsigned)(m / 128),
                                 (unsigned)batch), dim3(256), 0, stream>>>(p);
  };
  auto gemmSmall = [&](GemmBF p, int m, int n, int batch) {   // 128x64 tile
    gemm_bf16<4, 2, 2, 2><<<dim3((unsigned)(n / 64), (unsigned)(m / 128),
                                 (unsigned)batch), dim3(256), 0, stream>>>(p);
  };
  // projection: A[M][K] x Wt[N][K] (+bias,relu) -> bf16 shadow and/or fp32
  auto proj = [&](const u16* A, int lda, const u16* Wt, const float* bias,
                  int n, int k, int relu,
                  float* C, u16* Csh, int shMode, int ldsh) {
    GemmBF p{A, Wt, bias, C, Csh, M, n, k, lda, k, n, ldsh,
             0,0,0,0,0,0,0,0, 1, 1.0f, relu, shMode, C != nullptr};
    gemmBig(p, M, n, 1);
  };
  auto ln = [&](float* io, const float* add, const float* g, const float* b, u16* sh) {
    add_layernorm<<<dim3(M), dim3(256), 0, stream>>>(io, add, g, b, io, sh, D_MODEL);
  };

  const long long sBH = (long long)SEQ * D_MODEL;     // per-b stride in act rows
  const long long sSS = (long long)SEQ * SEQ;

  auto attn = [&](const u16* XqH, const u16* XkvH, const int* keyTok, int causal,
                  const u16* Wq, const float* bq, const u16* Wk, const float* bk,
                  const u16* Wv, const float* bv, const u16* Wo, const float* bo,
                  float* io, u16* ioh, const float* lng, const float* lnb) {
    proj(XqH,  D_MODEL, Wq, bq, D_MODEL, D_MODEL, 0, nullptr, Qh, 1, D_MODEL);
    proj(XkvH, D_MODEL, Wk, bk, D_MODEL, D_MODEL, 0, nullptr, Kh, 1, D_MODEL);
    proj(XkvH, D_MODEL, Wv, bv, D_MODEL, D_MODEL, 0, nullptr, Vt, 2, M);
    // scores[b,h] = 1/8 * Q_bh(512x64) x K_bh^T  (both K-contiguous bf16)
    {
      GemmBF p{Qh, Kh, nullptr, SC, nullptr, SEQ, SEQ, D_HEAD,
               D_MODEL, D_MODEL, SEQ, 0,
               sBH, D_HEAD, sBH, D_HEAD,
               (long long)N_HEAD * sSS, sSS, 0, 0,
               N_HEAD, 0.125f, 0, 0, 1};
      gemmBig(p, SEQ, SEQ, NB * N_HEAD);
    }
    softmax_rows<<<dim3(NB * N_HEAD * SEQ / 8), dim3(256), 0, stream>>>(
        SC, SCh, keyTok, causal, N_HEAD, SEQ, SEQ);
    // AO[b,:,h*64..] = P(512x512) x V(512x64); B = Vt[dk][pos]
    {
      GemmBF p{SCh, Vt, nullptr, nullptr, AOh, SEQ, D_HEAD, SEQ,
               SEQ, M, 0, D_MODEL,
               (long long)N_HEAD * sSS, sSS,
               (long long)SEQ, (long long)D_HEAD * M,
               0, 0, sBH, D_HEAD,
               N_HEAD, 1.0f, 0, 1, 0};
      gemmSmall(p, SEQ, D_HEAD, NB * N_HEAD);
    }
    proj(AOh, D_MODEL, Wo, bo, D_MODEL, D_MODEL, 0, Pb, nullptr, 0, 0);
    ln(io, Pb, lng, lnb, ioh);
  };
  auto ffn = [&](float* io, u16* ioh, const u16* W1t, const float* b1,
                 const u16* W2t, const float* b2, const float* lng, const float* lnb) {
    proj(ioh, D_MODEL, W1t, b1, D_FF, D_MODEL, 1, nullptr, Hh, 1, D_FF);
    proj(Hh, D_FF, W2t, b2, D_MODEL, D_FF, 0, Pb, nullptr, 0, 0);
    ln(io, Pb, lng, lnb, ioh);
  };

  // ---- embeddings (+PE); sqrt(D)=32 ----
  int total = M * D_MODEL;
  embed_pe<<<dim3(total / 256), dim3(256), 0, stream>>>(src, src_emb, X, Xh, SEQ, D_MODEL, 32.0f);
  embed_pe<<<dim3(total / 256), dim3(256), 0, stream>>>(tgt, tgt_emb, Y, Yh, SEQ, D_MODEL, 32.0f);

  // ---- encoder ----
  for (int i = 0; i < N_LAYER; ++i) {
    size_t oDD = (size_t)i * D_MODEL * D_MODEL;
    size_t oD  = (size_t)i * D_MODEL;
    size_t oDF = (size_t)i * D_MODEL * D_FF;
    size_t oF  = (size_t)i * D_FF;
    attn(Xh, Xh, src, 0,
         teWq + oDD, ebq + oD, teWk + oDD, ebk + oD,
         teWv + oDD, ebv + oD, teWo + oDD, ebo + oD,
         X, Xh, eln1g + oD, eln1b + oD);
    ffn(X, Xh, teW1 + oDF, eb1 + oF, teW2 + oDF, eb2 + oD, eln2g + oD, eln2b + oD);
  }

  // ---- decoder ----
  for (int i = 0; i < N_LAYER; ++i) {
    size_t oDD = (size_t)i * D_MODEL * D_MODEL;
    size_t oD  = (size_t)i * D_MODEL;
    size_t oDF = (size_t)i * D_MODEL * D_FF;
    size_t oF  = (size_t)i * D_FF;
    attn(Yh, Yh, tgt, 1,
         tdWq + oDD, dbq + oD, tdWk + oDD, dbk + oD,
         tdWv + oDD, dbv + oD, tdWo + oDD, dbo + oD,
         Y, Yh, dln1g + oD, dln1b + oD);
    attn(Yh, Xh, src, 0,
         tcWq + oDD, cbq + oD, tcWk + oDD, cbk + oD,
         tcWv + oDD, cbv + oD, tcWo + oDD, cbo + oD,
         Y, Yh, dln2g + oD, dln2b + oD);
    ffn(Y, Yh, tdW1 + oDF, db1 + oF, tdW2 + oDF, db2 + oD, dln3g + oD, dln3b + oD);
  }

  // ---- output projection: Yh[2048][1024] x outWt[32000][1024] + out_b ----
  {
    GemmBF p{Yh, toutW, outB, (float*)d_out, nullptr, M, VOCAB, D_MODEL,
             D_MODEL, D_MODEL, VOCAB, 0,
             0,0,0,0,0,0,0,0, 1, 1.0f, 0, 0, 1};
    gemmBig(p, M, VOCAB, 1);
  }
}
